// KANLayer_70866960384492
// MI455X (gfx1250) — compile-verified, matrix operand order checked by reading, not run on hardware
//
#include <hip/hip_runtime.h>
#include <stdint.h>

typedef _Float16 h8  __attribute__((ext_vector_type(8)));
typedef _Float16 h16 __attribute__((ext_vector_type(16)));
typedef float    f8  __attribute__((ext_vector_type(8)));

#define KTOT    2304      // 256 (gelu/base) + 256*8 (spline bases)
#define NSTEPS  72        // 2304 / 32
#define MTILE   64

// LDS layout (byte offsets into dynamic shared memory)
#define XT_OFF   0u         // x transposed [256 feat][65 rows] f32 = 66560 B (reused as Y [64][257] f32)
#define A_OFF    66560u     // A tile [64][40] f16 = 5120 B (32 used + 8 pad per row)
#define B0_OFF   71680u     // B tile buf0 [256][40] f16 = 20480 B
#define B1_OFF   92160u     // B tile buf1
#define PS_OFF   112640u    // 256 f32 partial sums
#define PSS_OFF  113664u    // 256 f32 partial sumsq
#define MU_OFF   114688u    // 64 f32
#define RS_OFF   114944u    // 64 f32
#define SMEM_BYTES 115200u

extern __shared__ __align__(16) char smem[];

// Load one 16x16x32 f16 fragment from an LDS row-major (K-contiguous) tile.
// Per CDNA5 ISA 16-bit A/B layout: lanes 0-15 hold K[0..7],K[16..23];
// lanes 16-31 hold K[8..15],K[24..31]  (half = lane>>4).
__device__ __forceinline__ h16 frag_ld(const _Float16* row, int half) {
  h8 lo = *(const h8*)(row + half * 8);
  h8 hi = *(const h8*)(row + 16 + half * 8);
  h16 r;
#pragma unroll
  for (int e = 0; e < 8; ++e) { r[e] = lo[e]; r[e + 8] = hi[e]; }
  return r;
}

// Async copy 64 B of one weight row (K-chunk) from global into LDS (CDNA5 path).
__device__ __forceinline__ void issue_btile(const char* g_row, char* lds_row) {
#pragma unroll
  for (int u = 0; u < 4; ++u) {
    unsigned l = (unsigned)(uintptr_t)(lds_row + u * 16);
    const void* g = (const void*)(g_row + u * 16);
    asm volatile("global_load_async_to_lds_b128 %0, %1, off"
                 :: "v"(l), "v"(g) : "memory");
  }
}

__device__ __forceinline__ void wait_async(int more_pending) {
  if (more_pending) asm volatile("s_wait_asynccnt 4" ::: "memory");
  else              asm volatile("s_wait_asynccnt 0" ::: "memory");
}

// -------- pack [base_weight | spline_weight] -> f16 Wp[256][2304] ----------
__global__ void kan_pack_w(const float* __restrict__ bw,
                           const float* __restrict__ sw,
                           _Float16* __restrict__ Wp) {
  const int o = blockIdx.x;
  for (int k = threadIdx.x; k < KTOT; k += 256) {
    float v = (k < 256) ? bw[o * 256 + k] : sw[(size_t)o * 2048 + (k - 256)];
    Wp[(size_t)o * KTOT + k] = (_Float16)v;
  }
}

// -------- fused: [gelu(x)|bases(x)] @ Wp^T -> LayerNorm -> PReLU -----------
__global__ __launch_bounds__(256) void kan_fused_wmma(
    const float* __restrict__ x, const float* __restrict__ grid,
    const _Float16* __restrict__ Wp, const float* __restrict__ gamma,
    const float* __restrict__ beta, const float* __restrict__ prelu_a,
    float* __restrict__ out) {
  const int t    = threadIdx.x;
  const int lane = t & 31;
  const int wave = t >> 5;
  const int row0 = blockIdx.x * MTILE;

  float*    XT  = (float*)(smem + XT_OFF);
  _Float16* At  = (_Float16*)(smem + A_OFF);
  _Float16* Bt0 = (_Float16*)(smem + B0_OFF);
  _Float16* Bt1 = (_Float16*)(smem + B1_OFF);

  // ---- stage x tile (64 rows x 256 feat) transposed into LDS ----
#pragma unroll
  for (int c = 0; c < 16; ++c) {
    int idx = c * 256 + t;            // float4 index, 4096 total
    int r   = idx >> 6;
    int c4  = idx & 63;
    float4 v = ((const float4*)(x + (size_t)(row0 + r) * 256))[c4];
    XT[(c4 * 4 + 0) * 65 + r] = v.x;
    XT[(c4 * 4 + 1) * 65 + r] = v.y;
    XT[(c4 * 4 + 2) * 65 + r] = v.z;
    XT[(c4 * 4 + 3) * 65 + r] = v.w;
  }

  const float kn0 = grid[0];
  const float ih  = 1.0f / (grid[1] - grid[0]);   // uniform knot spacing

  const char* gB   = (const char*)Wp + (size_t)t * (KTOT * 2);  // this thread's weight row
  char* ldsB0 = (char*)smem + B0_OFF + t * 80;
  char* ldsB1 = (char*)smem + B1_OFF + t * 80;

  issue_btile(gB, ldsB0);   // prefetch K-step 0

  __syncthreads();          // XT ready

  // wave -> 2 M-tiles x 4 O-tiles (2 A-frags + 4 B-frags feed 8 WMMAs)
  const int mg   = wave & 1;            // M group: rows mg*32 .. mg*32+31
  const int oq   = wave >> 1;           // O quarter: cols oq*64 .. oq*64+63
  const int half = lane >> 4;
  const int ln15 = lane & 15;

  f8 acc[8] = {};   // acc[a*4+j]: M-tile (2*mg+a), O-tile (oq*4+j)

  for (int ks = 0; ks < NSTEPS; ++ks) {
    const int k0   = ks * 32;
    const int cur  = ks & 1;
    const int more = (ks + 1 < NSTEPS);
    if (more) issue_btile(gB + (size_t)(k0 + 32) * 2, cur ? ldsB0 : ldsB1);

    // ---- build A tile [64 rows][32 K] f16 on the fly ----
    const int m = t & 63;
    const int q = t >> 6;
    if (k0 < 256) {                       // gelu(x) region
      h8 av;
#pragma unroll
      for (int e = 0; e < 8; ++e) {
        float xv = XT[(k0 + q * 8 + e) * 65 + m];
        float g  = 0.5f * xv * (1.0f + erff(xv * 0.70710678118654752f));
        av[e] = (_Float16)g;
      }
      *(h8*)(At + m * 40 + q * 8) = av;
    } else {
      // Cubic B-spline on uniform grid: only 4 of 8 bases are nonzero.
      // Cardinal weights for cell c, frac f; scatter via dynamic ds_store_b16
      // (out-of-range basis indices land in the row's 8-element pad).
      const int feat = ((k0 - 256) >> 3) + q;
      const float xv = XT[feat * 65 + m];
      const float u  = (xv - kn0) * ih;
      const float cf = floorf(u);
      const float f  = u - cf;
      const int   c  = (int)cf;
      const float f2 = f * f, f3 = f2 * f;
      const float omf = 1.0f - f;
      const float s6  = 1.0f / 6.0f;
      float w[4];
      w[0] = omf * omf * omf * s6;                 // basis c-3
      w[1] = (4.0f - 6.0f * f2 + 3.0f * f3) * s6;  // basis c-2
      w[2] = (1.0f + 3.0f * (f + f2 - f3)) * s6;   // basis c-1
      w[3] = f3 * s6;                              // basis c
      h8 z;
#pragma unroll
      for (int e = 0; e < 8; ++e) z[e] = (_Float16)0.0f;
      *(h8*)(At + m * 40 + q * 8) = z;             // zero the 8-basis slot
      const int jb = c - 3;
#pragma unroll
      for (int i = 0; i < 4; ++i) {
        int j   = jb + i;
        int off = ((unsigned)j < 8u) ? (q * 8 + j) : 34;  // 34 -> row pad
        At[m * 40 + off] = (_Float16)w[i];
      }
    }

    wait_async(more);        // current B buffer landed in LDS
    __syncthreads();

    const _Float16* Bc = cur ? Bt1 : Bt0;
    h16 af[2];
#pragma unroll
    for (int a = 0; a < 2; ++a)
      af[a] = frag_ld(At + (mg * 32 + a * 16 + ln15) * 40, half);
    h16 bf[4];
#pragma unroll
    for (int j = 0; j < 4; ++j)
      bf[j] = frag_ld(Bc + (oq * 64 + j * 16 + ln15) * 40, half);
#pragma unroll
    for (int a = 0; a < 2; ++a)
#pragma unroll
      for (int j = 0; j < 4; ++j)
        acc[a * 4 + j] = __builtin_amdgcn_wmma_f32_16x16x32_f16(
            false, af[a], false, bf[j], (short)0, acc[a * 4 + j], false, false);
    __syncthreads();
  }

  // ---- epilogue: y tile -> LDS, LayerNorm + PReLU, store ----
  float* Y = (float*)(smem + XT_OFF);     // reuse XT region, stride 257
#pragma unroll
  for (int a = 0; a < 2; ++a) {
    const int mbase = mg * 32 + a * 16 + half * 8;
#pragma unroll
    for (int j = 0; j < 4; ++j) {
      const int o = oq * 64 + j * 16 + ln15;
#pragma unroll
      for (int r = 0; r < 8; ++r) Y[(mbase + r) * 257 + o] = acc[a * 4 + j][r];
    }
  }
  float* PS  = (float*)(smem + PS_OFF);
  float* PSS = (float*)(smem + PSS_OFF);
  float* MUa = (float*)(smem + MU_OFF);
  float* RSa = (float*)(smem + RS_OFF);
  __syncthreads();

  const int r  = t & 63;
  const int q4 = t >> 6;
  {
    float s = 0.f, ss = 0.f;
#pragma unroll
    for (int c = 0; c < 64; ++c) {
      float v = Y[r * 257 + q4 * 64 + c];
      s += v; ss += v * v;
    }
    PS [q4 * 64 + r] = s;
    PSS[q4 * 64 + r] = ss;
  }
  __syncthreads();
  if (t < 64) {
    float s  = PS [t] + PS [64 + t] + PS [128 + t] + PS [192 + t];
    float ss = PSS[t] + PSS[64 + t] + PSS[128 + t] + PSS[192 + t];
    float mu  = s * (1.0f / 256.0f);
    float var = ss * (1.0f / 256.0f) - mu * mu;
    MUa[t] = mu;
    RSa[t] = rsqrtf(var + 1e-5f);
  }
  __syncthreads();

  const float ap = prelu_a[0];
  const float mu = MUa[r];
  const float rs = RSa[r];
  float4* orow = (float4*)(out + (size_t)(row0 + r) * 256);
#pragma unroll
  for (int c4 = 0; c4 < 16; ++c4) {
    const int col = q4 * 64 + c4 * 4;
    float vv[4];
#pragma unroll
    for (int e = 0; e < 4; ++e) {
      float yv = (Y[r * 257 + col + e] - mu) * rs * gamma[col + e] + beta[col + e];
      vv[e] = (yv >= 0.f) ? yv : ap * yv;
    }
    float4 v; v.x = vv[0]; v.y = vv[1]; v.z = vv[2]; v.w = vv[3];
    orow[col >> 2] = v;
  }
}

extern "C" void kernel_launch(void* const* d_in, const int* in_sizes, int n_in,
                              void* d_out, int out_size, void* d_ws, size_t ws_size,
                              hipStream_t stream) {
  (void)n_in; (void)out_size; (void)ws_size;
  const float* x     = (const float*)d_in[0];
  const float* grid  = (const float*)d_in[1];
  const float* bw    = (const float*)d_in[2];
  const float* sw    = (const float*)d_in[3];
  const float* gamma = (const float*)d_in[4];
  const float* beta  = (const float*)d_in[5];
  const float* pa    = (const float*)d_in[6];
  float* out = (float*)d_out;
  const int N = in_sizes[0] / 256;

  _Float16* Wp = (_Float16*)d_ws;   // needs 256*2304*2 = 1179648 B of scratch
  kan_pack_w<<<256, 256, 0, stream>>>(bw, sw, Wp);

  (void)hipFuncSetAttribute((const void*)kan_fused_wmma,
                            hipFuncAttributeMaxDynamicSharedMemorySize,
                            (int)SMEM_BYTES);
  kan_fused_wmma<<<N / MTILE, 256, SMEM_BYTES, stream>>>(
      x, grid, Wp, gamma, beta, pa, out);
}
